// ActorCritic_88983132439002
// MI455X (gfx1250) — matrix operand order repaired and checked by reading, not compiled, not run on hardware
//
#include <hip/hip_runtime.h>

// ---------------- problem constants ----------------
#define B_      1024
#define T_      256
#define F_      128
#define H_      128
#define G4      512      // 4*H
#define KK      256      // F+H
#define BT      32       // batch rows per workgroup
#define NWG     (B_/BT)  // 32 workgroups
#define THREADS 256      // 8 wave32 waves
#define SA      264      // LDS row stride (bf16) for activations (pad vs 64 banks)
#define SW      264      // LDS row stride (bf16) for weights (N-major)
#define SV      260      // LDS row stride (f32) for head activations
#define D_      256
#define A_      8
#define OUTW    17       // mean(8) + std(8) + value(1)

#define SMEM_BYTES (BT*SA*2 + G4*SW*2)   // 16896 + 270336 = 287232 B < 320KB WGP LDS

typedef __bf16 bf8v  __attribute__((ext_vector_type(8)));
typedef __bf16 bf16v __attribute__((ext_vector_type(16)));
typedef float  f8v   __attribute__((ext_vector_type(8)));

__device__ __forceinline__ unsigned short f2bf(float f) {
  union { float f; unsigned u; } x; x.f = f;
  unsigned r = 0x7FFFu + ((x.u >> 16) & 1u);      // round-to-nearest-even
  return (unsigned short)((x.u + r) >> 16);
}
__device__ __forceinline__ unsigned pk2(float lo, float hi) {
  return (unsigned)f2bf(lo) | ((unsigned)f2bf(hi) << 16);
}
__device__ __forceinline__ f8v f8zero() {
  f8v z;
#pragma unroll
  for (int i = 0; i < 8; ++i) z[i] = 0.f;
  return z;
}
__device__ __forceinline__ float sigf(float x) { return 1.0f / (1.0f + __expf(-x)); }

// A-fragment, 16x32 bf16: lanes 0-15 -> K kb+0..15,
// lanes 16-31 -> K kb+16..31; row m = lane&15 (rowp pre-offset by row).
__device__ __forceinline__ bf16v ldA(const __bf16* rowp, int kb, int half) {
  bf8v lo = *(const bf8v*)(rowp + kb + half * 16);
  bf8v hi = *(const bf8v*)(rowp + kb + half * 16 + 8);
  bf16v r;
#pragma unroll
  for (int e = 0; e < 8; ++e) { r[e] = lo[e]; r[e + 8] = hi[e]; }
  return r;
}
// B-fragment, 32x16 bf16: lane holds col n = lane&15, 16 contiguous K at kb+half*16.
// Weight storage is N-major [N][K], so this is one 32-byte contiguous read.
__device__ __forceinline__ bf16v ldB(const __bf16* rowp, int kb, int half) {
  bf8v lo = *(const bf8v*)(rowp + kb + half * 16);
  bf8v hi = *(const bf8v*)(rowp + kb + half * 16 + 8);
  bf16v r;
#pragma unroll
  for (int e = 0; e < 8; ++e) { r[e] = lo[e]; r[e + 8] = hi[e]; }
  return r;
}

// ---------------- weight prep: f32 [K][N] -> bf16 [N][K] ----------------
__global__ void prep_wt(const float* __restrict__ Wx, const float* __restrict__ Wh,
                        __bf16* __restrict__ WT) {
  int idx = blockIdx.x * 256 + threadIdx.x;          // 512*256 combined [Wx;Wh]^T
  if (idx >= G4 * KK) return;
  int n = idx >> 8, k = idx & 255;
  float v = (k < F_) ? Wx[k * G4 + n] : Wh[(k - F_) * G4 + n];
  *(unsigned short*)(WT + idx) = f2bf(v);
}
__global__ void prep_tr(const float* __restrict__ src, __bf16* __restrict__ dst,
                        int K, int N) {
  int idx = blockIdx.x * 256 + threadIdx.x;
  if (idx >= K * N) return;
  int n = idx / K, k = idx - n * K;
  *(unsigned short*)(dst + idx) = f2bf(src[k * N + n]);
}

// ---------------- head GEMM: [32 x K] (LDS bf16) @ [K x 256] (global bf16 N-major) ----------------
template <int KSTEPS, bool F32OUT>
__device__ __forceinline__ void head_gemm(const __bf16* sAbase, int colOff,
                                          const __bf16* __restrict__ Bg, int strideK,
                                          const float* __restrict__ bias,
                                          __bf16* dH, float* dV, int wid, int lane) {
  const int half = lane >> 4, col = lane & 15;
#pragma unroll
  for (int ct = 0; ct < 2; ++ct) {
    const int n0 = (wid * 2 + ct) * 16;
    const __bf16* Brow = Bg + (size_t)(n0 + col) * strideK;
    f8v acc[2];
#pragma unroll
    for (int rb = 0; rb < 2; ++rb) acc[rb] = f8zero();
    for (int k = 0; k < KSTEPS; ++k) {
      const int kb = k * 32;
      bf16v b = ldB(Brow, kb, half);
#pragma unroll
      for (int rb = 0; rb < 2; ++rb) {
        bf16v a = ldA(sAbase + (rb * 16 + col) * SA + colOff, kb, half);
        acc[rb] = __builtin_amdgcn_wmma_f32_16x16x32_bf16(false, a, false, b,
                                                          (short)0, acc[rb], false, false);
      }
    }
    const float bv = bias[n0 + col];
#pragma unroll
    for (int rb = 0; rb < 2; ++rb)
#pragma unroll
      for (int v = 0; v < 8; ++v) {
        float t = tanhf(acc[rb][v] + bv);
        int row = rb * 16 + v + half * 8;
        if (F32OUT) dV[row * SV + n0 + col] = t;
        else *(unsigned short*)(dH + row * SA + n0 + col) = f2bf(t);
      }
  }
}

// ---------------- fused LSTM + actor/critic heads ----------------
__global__ __launch_bounds__(THREADS, 1)
void lstm_ac_kernel(const float* __restrict__ h,   const float* __restrict__ bh,
                    const float* __restrict__ ba1, const float* __restrict__ ba2,
                    const float* __restrict__ Wa3, const float* __restrict__ ba3,
                    const float* __restrict__ lstd,
                    const float* __restrict__ bc1, const float* __restrict__ bc2,
                    const float* __restrict__ Wc3, const float* __restrict__ bc3,
                    const __bf16* __restrict__ WT,
                    const __bf16* __restrict__ WaT1, const __bf16* __restrict__ WaT2,
                    const __bf16* __restrict__ WcT1, const __bf16* __restrict__ WcT2,
                    float* __restrict__ out) {
  extern __shared__ char smem[];
  __bf16* sA = (__bf16*)smem;                       // [BT][SA]  (x_t | h_{t-1}) bf16
  __bf16* sW = (__bf16*)(smem + BT * SA * 2);       // [512][SW] combined LSTM weights
  __bf16* sH = sW;                                  // heads reuse (a1, bf16)
  float*  sV = (float*)(smem + 2 * (BT * SA * 2));  // heads reuse (a2, f32)

  const int tid = threadIdx.x;
  const int wid = tid >> 5;
  const int lane = tid & 31;
  const int half = lane >> 4;
  const int col = lane & 15;
  const int b0 = blockIdx.x * BT;
  const int n0 = wid * 16;                          // H-column block owned by this wave

  // stage combined weight [512][256] bf16 into LDS (16B vector copies)
  for (int i = tid; i < (G4 * KK) / 8; i += THREADS) {
    int n = i >> 5, k = (i & 31) * 8;
    *(uint4*)(sW + n * SW + k) = ((const uint4*)WT)[i];
  }
  // h_{-1} = 0
  for (int i = tid; i < (BT * H_) / 8; i += THREADS) {
    int row = i >> 4, k = (i & 15) * 8;
    uint4 z; z.x = z.y = z.z = z.w = 0u;
    *(uint4*)(sA + row * SA + F_ + k) = z;
  }

  // per-lane gate biases (gate order i,f,g,o)
  const float bI = bh[0 * H_ + n0 + col];
  const float bF = bh[1 * H_ + n0 + col];
  const float bG = bh[2 * H_ + n0 + col];
  const float bO = bh[3 * H_ + n0 + col];

  const __bf16* BrowG[4];
#pragma unroll
  for (int g = 0; g < 4; ++g) BrowG[g] = sW + (g * H_ + n0 + col) * SW;

  float cReg[2][8];
#pragma unroll
  for (int rb = 0; rb < 2; ++rb)
#pragma unroll
    for (int v = 0; v < 8; ++v) cReg[rb][v] = 0.f;

  const int xrow = tid >> 3;   // 32 rows, 8 threads/row
  const int xpart = tid & 7;   // 16 f32 each
  const float* xsrc0 = h + (size_t)(b0 + xrow) * T_ * F_ + xpart * 16;

  // ---- software pipeline: prefetch x(0) into registers ----
  float4 p0, p1, p2, p3;
  {
    const float4* src = (const float4*)xsrc0;
    p0 = src[0]; p1 = src[1]; p2 = src[2]; p3 = src[3];
  }

#pragma unroll 1
  for (int t = 0; t < T_; ++t) {
    // ---- convert prefetched x_t regs -> bf16 into sA cols [0,128) ----
    uint4 d0, d1;
    d0.x = pk2(p0.x, p0.y); d0.y = pk2(p0.z, p0.w);
    d0.z = pk2(p1.x, p1.y); d0.w = pk2(p1.z, p1.w);
    d1.x = pk2(p2.x, p2.y); d1.y = pk2(p2.z, p2.w);
    d1.z = pk2(p3.x, p3.y); d1.w = pk2(p3.z, p3.w);
    uint4* dst = (uint4*)(sA + xrow * SA + xpart * 16);
    dst[0] = d0; dst[1] = d1;

    // ---- issue global loads for x_{t+1}: latency hidden behind this step ----
    if (t + 1 < T_) {
      const float4* src = (const float4*)(xsrc0 + (size_t)(t + 1) * F_);
      p0 = src[0]; p1 = src[1]; p2 = src[2]; p3 = src[3];
    }
    __syncthreads();  // x_t and h_{t-1} visible

    // ---- Z = [x|h] @ W : per wave 2 row-blocks x 4 gates, K=256 ----
    f8v acc[4][2];
#pragma unroll
    for (int g = 0; g < 4; ++g)
#pragma unroll
      for (int rb = 0; rb < 2; ++rb) acc[g][rb] = f8zero();

    for (int k = 0; k < 8; ++k) {
      const int kb = k * 32;
      bf16v bg0 = ldB(BrowG[0], kb, half);
      bf16v bg1 = ldB(BrowG[1], kb, half);
      bf16v bg2 = ldB(BrowG[2], kb, half);
      bf16v bg3 = ldB(BrowG[3], kb, half);
#pragma unroll
      for (int rb = 0; rb < 2; ++rb) {
        bf16v a = ldA(sA + (rb * 16 + col) * SA, kb, half);
        acc[0][rb] = __builtin_amdgcn_wmma_f32_16x16x32_bf16(false, a, false, bg0, (short)0, acc[0][rb], false, false);
        acc[1][rb] = __builtin_amdgcn_wmma_f32_16x16x32_bf16(false, a, false, bg1, (short)0, acc[1][rb], false, false);
        acc[2][rb] = __builtin_amdgcn_wmma_f32_16x16x32_bf16(false, a, false, bg2, (short)0, acc[2][rb], false, false);
        acc[3][rb] = __builtin_amdgcn_wmma_f32_16x16x32_bf16(false, a, false, bg3, (short)0, acc[3][rb], false, false);
      }
    }
    __syncthreads();  // all reads of sA done before h update

    // ---- gate math (f32), update c (regs) and h (LDS bf16) ----
#pragma unroll
    for (int rb = 0; rb < 2; ++rb)
#pragma unroll
      for (int v = 0; v < 8; ++v) {
        float zi = acc[0][rb][v] + bI;
        float zf = acc[1][rb][v] + bF;
        float zg = acc[2][rb][v] + bG;
        float zo = acc[3][rb][v] + bO;
        float c = sigf(zf) * cReg[rb][v] + sigf(zi) * tanhf(zg);
        cReg[rb][v] = c;
        float hn = sigf(zo) * tanhf(c);
        int row = rb * 16 + v + half * 8;
        *(unsigned short*)(sA + row * SA + F_ + n0 + col) = f2bf(hn);
      }
  }
  __syncthreads();

  // ---- actor head ----
  head_gemm<4, false>(sA, F_, WaT1, 128, ba1, sH, sV, wid, lane);
  __syncthreads();
  head_gemm<8, true >(sH, 0,  WaT2, 256, ba2, sH, sV, wid, lane);
  __syncthreads();
  {
    int row = tid >> 3, a = tid & 7;                 // 32*8 == 256 threads
    float s = ba3[a];
    for (int k = 0; k < D_; ++k) s += sV[row * SV + k] * Wa3[k * A_ + a];
    float* o = out + (size_t)(b0 + row) * OUTW;
    o[a]     = s;
    o[8 + a] = __expf(lstd[a]);
  }
  __syncthreads();

  // ---- critic head ----
  head_gemm<4, false>(sA, F_, WcT1, 128, bc1, sH, sV, wid, lane);
  __syncthreads();
  head_gemm<8, true >(sH, 0,  WcT2, 256, bc2, sH, sV, wid, lane);
  __syncthreads();
  if (tid < BT) {
    float s = bc3[0];
    for (int k = 0; k < D_; ++k) s += sV[tid * SV + k] * Wc3[k];
    out[(size_t)(b0 + tid) * OUTW + 16] = s;
  }
}

// ---------------- host launch ----------------
extern "C" void kernel_launch(void* const* d_in, const int* in_sizes, int n_in,
                              void* d_out, int out_size, void* d_ws, size_t ws_size,
                              hipStream_t stream) {
  (void)in_sizes; (void)n_in; (void)out_size; (void)ws_size;
  const float* h    = (const float*)d_in[0];
  const float* Wx   = (const float*)d_in[1];
  const float* Wh   = (const float*)d_in[2];
  const float* bh   = (const float*)d_in[3];
  const float* Wa1  = (const float*)d_in[4];
  const float* ba1  = (const float*)d_in[5];
  const float* Wa2  = (const float*)d_in[6];
  const float* ba2  = (const float*)d_in[7];
  const float* Wa3  = (const float*)d_in[8];
  const float* ba3  = (const float*)d_in[9];
  const float* lstd = (const float*)d_in[10];
  const float* Wc1  = (const float*)d_in[11];
  const float* bc1  = (const float*)d_in[12];
  const float* Wc2  = (const float*)d_in[13];
  const float* bc2  = (const float*)d_in[14];
  const float* Wc3  = (const float*)d_in[15];
  const float* bc3  = (const float*)d_in[16];
  float* out = (float*)d_out;

  __bf16* WT   = (__bf16*)d_ws;          // [512][256]
  __bf16* WaT1 = WT   + 512 * 256;       // [256][128]
  __bf16* WaT2 = WaT1 + 256 * 128;       // [256][256]
  __bf16* WcT1 = WaT2 + 256 * 256;       // [256][128]
  __bf16* WcT2 = WcT1 + 256 * 128;       // [256][256]

  prep_wt<<<512, 256, 0, stream>>>(Wx, Wh, WT);
  prep_tr<<<128, 256, 0, stream>>>(Wa1, WaT1, 128, 256);
  prep_tr<<<256, 256, 0, stream>>>(Wa2, WaT2, 256, 256);
  prep_tr<<<128, 256, 0, stream>>>(Wc1, WcT1, 128, 256);
  prep_tr<<<256, 256, 0, stream>>>(Wc2, WcT2, 256, 256);

  hipFuncSetAttribute(reinterpret_cast<const void*>(lstm_ac_kernel),
                      hipFuncAttributeMaxDynamicSharedMemorySize, SMEM_BYTES);
  lstm_ac_kernel<<<NWG, THREADS, SMEM_BYTES, stream>>>(
      h, bh, ba1, ba2, Wa3, ba3, lstd, bc1, bc2, Wc3, bc3,
      WT, WaT1, WaT2, WcT1, WcT2, out);
}
